// POSTagger_14542759264570
// MI455X (gfx1250) — compile-verified, matrix operand order checked by reading, not run on hardware
//
#include <hip/hip_runtime.h>
#include <hip/hip_bf16.h>

// ---- problem constants ----
#define BB   512            // batch
#define SS   64             // seq len
#define WW   12             // chars per word
#define CV   100
#define CE   10
#define FF   32             // conv filters
#define WE   250
#define HH   250
#define HP   256            // padded hidden (per gate)
#define IN0  282            // WE + FF
#define INP0 288            // padded layer-0 x width (9 k-blocks of 32)
#define K1   512            // layer-1 concat K (256 x + 256 h)
#define NG   1024           // 4 gates * 256
#define TT   45             // tagset
#define ROWS (BB*SS)        // 32768 flat rows

typedef __attribute__((ext_vector_type(16))) __bf16        v16bf;
typedef __attribute__((ext_vector_type(8)))  float         v8f;
typedef __attribute__((ext_vector_type(4)))  unsigned int  u32x4;

union FragU { v16bf v; u32x4 q[2]; };

// One 16x32 bf16 A/B fragment slice per lane (ISA 16-bit layout):
// half 0 -> K = base+0..7, base+16..23 ; half 1 -> +8..15, +24..31
__device__ __forceinline__ v16bf load_frag(const __bf16* p, int kbase, int half) {
    FragU f;
    f.q[0] = *(const u32x4*)(p + kbase + half * 8);
    f.q[1] = *(const u32x4*)(p + kbase + 16 + half * 8);
    return f.v;
}

__device__ __forceinline__ float sigm(float x) { return 1.0f / (1.0f + __expf(-x)); }

// Depth-2 software-pipelined 4-gate GEMM slice: acc_g += A(16 x KB*32) * B_g.
// Three rotating fragment sets; loads issue 2 k-iterations (8 WMMAs) ahead
// of their consuming WMMA.
template<int KB>
__device__ __forceinline__ void gemm4(const __bf16* __restrict__ aRow,
                                      const __bf16* __restrict__ w0,
                                      const __bf16* __restrict__ w1,
                                      const __bf16* __restrict__ w2,
                                      const __bf16* __restrict__ w3,
                                      int half,
                                      v8f& acc0, v8f& acc1, v8f& acc2, v8f& acc3)
{
    v16bf a[3], b0[3], b1[3], b2[3], b3[3];
    a[0]  = load_frag(aRow, 0, half);
    b0[0] = load_frag(w0, 0, half);
    b1[0] = load_frag(w1, 0, half);
    b2[0] = load_frag(w2, 0, half);
    b3[0] = load_frag(w3, 0, half);
    if (KB > 1) {
        a[1]  = load_frag(aRow, 32, half);
        b0[1] = load_frag(w0, 32, half);
        b1[1] = load_frag(w1, 32, half);
        b2[1] = load_frag(w2, 32, half);
        b3[1] = load_frag(w3, 32, half);
    }
    #pragma unroll
    for (int kb = 0; kb < KB; ++kb) {
        const int cur = kb % 3;
        const int nxt = (kb + 2) % 3;
        if (kb + 2 < KB) {                    // prefetch 2 iterations ahead
            int kc = (kb + 2) * 32;
            a[nxt]  = load_frag(aRow, kc, half);
            b0[nxt] = load_frag(w0, kc, half);
            b1[nxt] = load_frag(w1, kc, half);
            b2[nxt] = load_frag(w2, kc, half);
            b3[nxt] = load_frag(w3, kc, half);
        }
        acc0 = __builtin_amdgcn_wmma_f32_16x16x32_bf16(false, a[cur], false, b0[cur], (short)0, acc0, false, false);
        acc1 = __builtin_amdgcn_wmma_f32_16x16x32_bf16(false, a[cur], false, b1[cur], (short)0, acc1, false, false);
        acc2 = __builtin_amdgcn_wmma_f32_16x16x32_bf16(false, a[cur], false, b2[cur], (short)0, acc2, false, false);
        acc3 = __builtin_amdgcn_wmma_f32_16x16x32_bf16(false, a[cur], false, b3[cur], (short)0, acc3, false, false);
    }
}

// ------------------------------------------------------------------
// Kernel 0: repack weights to bf16 padded layouts, combine biases
// ------------------------------------------------------------------
__global__ void prep_kernel(const float* __restrict__ wih0, const float* __restrict__ whh0,
                            const float* __restrict__ bih0, const float* __restrict__ bhh0,
                            const float* __restrict__ wih1, const float* __restrict__ whh1,
                            const float* __restrict__ bih1, const float* __restrict__ bhh1,
                            const float* __restrict__ dw,   const float* __restrict__ db,
                            __bf16* __restrict__ W0x, __bf16* __restrict__ W0h,
                            __bf16* __restrict__ W1,  __bf16* __restrict__ DW,
                            float* __restrict__ bias0, float* __restrict__ bias1,
                            float* __restrict__ DB)
{
    int tid = blockIdx.x * blockDim.x + threadIdx.x;
    int stride = gridDim.x * blockDim.x;

    for (int idx = tid; idx < NG * INP0; idx += stride) {
        int n = idx / INP0, col = idx % INP0;
        int g = n >> 8, j = n & 255;
        float v = (j < HH && col < IN0) ? wih0[(g * HH + j) * IN0 + col] : 0.f;
        W0x[idx] = (__bf16)v;
    }
    for (int idx = tid; idx < NG * HP; idx += stride) {
        int n = idx / HP, col = idx % HP;
        int g = n >> 8, j = n & 255;
        float v = (j < HH && col < HH) ? whh0[(g * HH + j) * HH + col] : 0.f;
        W0h[idx] = (__bf16)v;
    }
    for (int idx = tid; idx < NG * K1; idx += stride) {
        int n = idx / K1, col = idx % K1;
        int g = n >> 8, j = n & 255;
        float v = 0.f;
        if (j < HH) {
            if (col < HP) { if (col < HH) v = wih1[(g * HH + j) * HH + col]; }
            else { int kh = col - HP; if (kh < HH) v = whh1[(g * HH + j) * HH + kh]; }
        }
        W1[idx] = (__bf16)v;
    }
    for (int n = tid; n < NG; n += stride) {
        int g = n >> 8, j = n & 255;
        bias0[n] = (j < HH) ? bih0[g * HH + j] + bhh0[g * HH + j] : 0.f;
        bias1[n] = (j < HH) ? bih1[g * HH + j] + bhh1[g * HH + j] : 0.f;
    }
    for (int idx = tid; idx < 64 * HP; idx += stride) {
        int t = idx / HP, j = idx % HP;
        DW[idx] = (__bf16)((t < TT && j < HH) ? dw[t * HH + j] : 0.f);
    }
    for (int t = tid; t < 64; t += stride) DB[t] = (t < TT) ? db[t] : 0.f;
}

// ------------------------------------------------------------------
// Kernel 1: zero LSTM state (c buffers, step-0 h slots)
// ------------------------------------------------------------------
__global__ void zero_state(float* __restrict__ c0, float* __restrict__ c1,
                           __bf16* __restrict__ h1s0, __bf16* __restrict__ h2s0)
{
    int i = blockIdx.x * blockDim.x + threadIdx.x;
    int stride = gridDim.x * blockDim.x;
    for (int k = i; k < BB * HP; k += stride) {
        c0[k] = 0.f; c1[k] = 0.f;
        h1s0[k] = (__bf16)0.f; h2s0[k] = (__bf16)0.f;
    }
}

// ------------------------------------------------------------------
// Kernel 2: char embed -> conv1d(k=3,pad=1) -> max  ||  word embed
// ------------------------------------------------------------------
__global__ void embed_kernel(const int* __restrict__ ci, const int* __restrict__ wi,
                             const float* __restrict__ charEmb, const float* __restrict__ wordEmb,
                             const float* __restrict__ convW, const float* __restrict__ convB,
                             __bf16* __restrict__ emb)
{
    __shared__ float sCE[CV * CE];
    __shared__ float sCW[FF * CE * 3];
    __shared__ float sCB[FF];
    __shared__ int   sIdx[WW];

    int t = threadIdx.x;
    int bs = blockIdx.x;                       // flat (s*512 + b) row
    for (int i = t; i < CV * CE; i += 128) sCE[i] = charEmb[i];
    for (int i = t; i < FF * CE * 3; i += 128) sCW[i] = convW[i];
    if (t < FF) sCB[t] = convB[t];
    if (t < WW) sIdx[t] = ci[bs * WW + t];
    __syncthreads();

    __bf16* row = emb + (size_t)bs * INP0;
    if (t < FF) {
        float mx = -3.0e38f;
        for (int w = 0; w < WW; ++w) {
            float acc = sCB[t];
            #pragma unroll
            for (int tp = 0; tp < 3; ++tp) {
                int wc = w + tp - 1;
                if (wc >= 0 && wc < WW) {
                    const float* ce = &sCE[sIdx[wc] * CE];
                    const float* cw = &sCW[t * 30 + tp];
                    #pragma unroll
                    for (int c = 0; c < CE; ++c) acc += ce[c] * cw[c * 3];
                }
            }
            mx = fmaxf(mx, acc);
        }
        row[t] = (__bf16)mx;                   // cols 0..31
    } else {
        int widx = wi[bs];
        for (int c = t - FF; c < HP; c += 96) {
            float v = (c < WE) ? wordEmb[(size_t)widx * WE + c] : 0.f;
            row[FF + c] = (__bf16)v;           // cols 32..287 (282..287 zero)
        }
    }
}

// ------------------------------------------------------------------
// Kernel 3: layer-0 input projection (fully parallel, off critical path)
// xg0(32768 x 1024) = emb(32768 x 288) @ W0x^T + bias0, stored bf16
// ------------------------------------------------------------------
__global__ void __launch_bounds__(128, 1)
input_proj(const __bf16* __restrict__ emb, const __bf16* __restrict__ W0x,
           const float* __restrict__ bias0, __bf16* __restrict__ xg)
{
    int lane = threadIdx.x & 31;
    int wave = (blockIdx.x << 2) + (threadIdx.x >> 5);   // 0..32767
    int half = lane >> 4, ln = lane & 15;
    int mTile = wave >> 4;                               // 0..2047
    int jTile = wave & 15;
    int m = mTile * 16 + ln;
    int j = jTile * 16 + ln;

    const __bf16* aRow = emb + (size_t)m * INP0;
    const __bf16* w0 = W0x + (size_t)(0 * HP + j) * INP0;
    const __bf16* w1 = W0x + (size_t)(1 * HP + j) * INP0;
    const __bf16* w2 = W0x + (size_t)(2 * HP + j) * INP0;
    const __bf16* w3 = W0x + (size_t)(3 * HP + j) * INP0;

    v8f a0 = {}, a1 = {}, a2 = {}, a3 = {};
    gemm4<INP0 / 32>(aRow, w0, w1, w2, w3, half, a0, a1, a2, a3);

    float bI = bias0[0 * HP + j], bF = bias0[1 * HP + j];
    float bG = bias0[2 * HP + j], bO = bias0[3 * HP + j];
    #pragma unroll
    for (int v = 0; v < 8; ++v) {
        size_t r = (size_t)(mTile * 16 + v + half * 8);
        __bf16* xr = xg + r * NG;
        xr[0 * HP + j] = (__bf16)(a0[v] + bI);
        xr[1 * HP + j] = (__bf16)(a1[v] + bF);
        xr[2 * HP + j] = (__bf16)(a2[v] + bG);
        xr[3 * HP + j] = (__bf16)(a3[v] + bO);
    }
}

// ------------------------------------------------------------------
// Kernel 4: fused LSTM step.  Launch s = 0..64:
//   blocks [0,128)   : layer 0, time step s     (skipped when s==64)
//   blocks [128,256) : layer 1, time step s-1   (skipped when s==0)
// All dependencies are cross-launch -> stream order is the recurrence.
// ------------------------------------------------------------------
__global__ void __launch_bounds__(128, 1)
lstm_fused(int s,
           const __bf16* __restrict__ xg0,
           __bf16* __restrict__ h1seq, __bf16* __restrict__ h2seq,
           const __bf16* __restrict__ W0h, const __bf16* __restrict__ W1,
           const float* __restrict__ bias1,
           float* __restrict__ c0, float* __restrict__ c1)
{
    const size_t slot = (size_t)BB * HP;
    int lane = threadIdx.x & 31;
    int half = lane >> 4, ln = lane & 15;

    if (blockIdx.x < 128) {
        // ---------------- layer 0, step s ----------------
        if (s >= SS) return;
        int wave = (blockIdx.x << 2) + (threadIdx.x >> 5);   // 0..511
        int mTile = wave >> 4, jTile = wave & 15;
        int m = mTile * 16 + ln;
        int j = jTile * 16 + ln;

        const __bf16* hRow = h1seq + (size_t)s * slot + (size_t)m * HP;
        const __bf16* w0 = W0h + (size_t)(0 * HP + j) * HP;
        const __bf16* w1 = W0h + (size_t)(1 * HP + j) * HP;
        const __bf16* w2 = W0h + (size_t)(2 * HP + j) * HP;
        const __bf16* w3 = W0h + (size_t)(3 * HP + j) * HP;

        v8f a0 = {}, a1 = {}, a2 = {}, a3 = {};
        gemm4<HP / 32>(hRow, w0, w1, w2, w3, half, a0, a1, a2, a3);

        __bf16* hOut = h1seq + (size_t)(s + 1) * slot;
        #pragma unroll
        for (int v = 0; v < 8; ++v) {
            int brow = mTile * 16 + v + half * 8;
            const __bf16* xr = xg0 + (size_t)(s * BB + brow) * NG;
            float xi = a0[v] + (float)xr[0 * HP + j];
            float xf = a1[v] + (float)xr[1 * HP + j];
            float xg = a2[v] + (float)xr[2 * HP + j];
            float xo = a3[v] + (float)xr[3 * HP + j];
            size_t cix = (size_t)brow * HP + j;
            float cn = sigm(xf) * c0[cix] + sigm(xi) * tanhf(xg);
            c0[cix] = cn;
            hOut[cix] = (__bf16)(sigm(xo) * tanhf(cn));
        }
    } else {
        // ---------------- layer 1, step s-1 ----------------
        if (s < 1) return;
        int t = s - 1;
        int wave = ((blockIdx.x - 128) << 2) + (threadIdx.x >> 5);   // 0..511
        int mTile = wave >> 4, jTile = wave & 15;
        int m = mTile * 16 + ln;
        int j = jTile * 16 + ln;

        const __bf16* xRow = h1seq + (size_t)(t + 1) * slot + (size_t)m * HP;  // h1[t]
        const __bf16* hRow = h2seq + (size_t)t * slot + (size_t)m * HP;
        const __bf16* w0 = W1 + (size_t)(0 * HP + j) * K1;
        const __bf16* w1 = W1 + (size_t)(1 * HP + j) * K1;
        const __bf16* w2 = W1 + (size_t)(2 * HP + j) * K1;
        const __bf16* w3 = W1 + (size_t)(3 * HP + j) * K1;

        v8f a0 = {}, a1 = {}, a2 = {}, a3 = {};
        gemm4<HP / 32>(xRow, w0, w1, w2, w3, half, a0, a1, a2, a3);                 // x part
        gemm4<HP / 32>(hRow, w0 + HP, w1 + HP, w2 + HP, w3 + HP, half, a0, a1, a2, a3); // h part

        float bI = bias1[0 * HP + j], bF = bias1[1 * HP + j];
        float bG = bias1[2 * HP + j], bO = bias1[3 * HP + j];
        __bf16* hOut = h2seq + (size_t)(t + 1) * slot;
        #pragma unroll
        for (int v = 0; v < 8; ++v) {
            int brow = mTile * 16 + v + half * 8;
            float xi = a0[v] + bI, xf = a1[v] + bF, xg = a2[v] + bG, xo = a3[v] + bO;
            size_t cix = (size_t)brow * HP + j;
            float cn = sigm(xf) * c1[cix] + sigm(xi) * tanhf(xg);
            c1[cix] = cn;
            hOut[cix] = (__bf16)(sigm(xo) * tanhf(cn));
        }
    }
}

// ------------------------------------------------------------------
// Kernel 5: dense tag projection: (32768x256) @ (256x48), 45 cols used
// ------------------------------------------------------------------
__global__ void __launch_bounds__(128, 1)
dense_kernel(const __bf16* __restrict__ H2seq,
             const __bf16* __restrict__ DW,
             const float* __restrict__ DB,
             float* __restrict__ out)
{
    int lane = threadIdx.x & 31;
    int wave = (blockIdx.x << 2) + (threadIdx.x >> 5);    // 0..2047 = mTile
    int half = lane >> 4, ln = lane & 15;
    int m = wave * 16 + ln;
    const __bf16* hRow = H2seq + (size_t)(BB + m) * HP;   // skip zero slot

    const __bf16* w0 = DW + (size_t)(0 * 16 + ln) * HP;
    const __bf16* w1 = DW + (size_t)(1 * 16 + ln) * HP;
    const __bf16* w2 = DW + (size_t)(2 * 16 + ln) * HP;

    v8f acc0 = {}, acc1 = {}, acc2 = {};
    v16bf a[3], b0[3], b1[3], b2[3];
    a[0]  = load_frag(hRow, 0, half);
    b0[0] = load_frag(w0, 0, half);
    b1[0] = load_frag(w1, 0, half);
    b2[0] = load_frag(w2, 0, half);
    a[1]  = load_frag(hRow, 32, half);
    b0[1] = load_frag(w0, 32, half);
    b1[1] = load_frag(w1, 32, half);
    b2[1] = load_frag(w2, 32, half);
    #pragma unroll
    for (int kb = 0; kb < HP / 32; ++kb) {
        const int cur = kb % 3;
        const int nxt = (kb + 2) % 3;
        if (kb + 2 < HP / 32) {
            int kc = (kb + 2) * 32;
            a[nxt]  = load_frag(hRow, kc, half);
            b0[nxt] = load_frag(w0, kc, half);
            b1[nxt] = load_frag(w1, kc, half);
            b2[nxt] = load_frag(w2, kc, half);
        }
        acc0 = __builtin_amdgcn_wmma_f32_16x16x32_bf16(false, a[cur], false, b0[cur], (short)0, acc0, false, false);
        acc1 = __builtin_amdgcn_wmma_f32_16x16x32_bf16(false, a[cur], false, b1[cur], (short)0, acc1, false, false);
        acc2 = __builtin_amdgcn_wmma_f32_16x16x32_bf16(false, a[cur], false, b2[cur], (short)0, acc2, false, false);
    }

    #pragma unroll
    for (int v = 0; v < 8; ++v) {
        int r = wave * 16 + v + half * 8;                 // output flat row (S,B) order
        float* orow = out + (size_t)r * TT;
        int t0 = ln, t1 = 16 + ln, t2 = 32 + ln;
        if (t0 < TT) orow[t0] = acc0[v] + DB[t0];
        if (t1 < TT) orow[t1] = acc1[v] + DB[t1];
        if (t2 < TT) orow[t2] = acc2[v] + DB[t2];
    }
}

// ------------------------------------------------------------------
static inline size_t alignup(size_t x) { return (x + 255) & ~(size_t)255; }

extern "C" void kernel_launch(void* const* d_in, const int* in_sizes, int n_in,
                              void* d_out, int out_size, void* d_ws, size_t ws_size,
                              hipStream_t stream)
{
    (void)in_sizes; (void)n_in; (void)out_size; (void)ws_size;
    const int*   char_idx = (const int*)  d_in[0];
    const int*   word_idx = (const int*)  d_in[1];
    const float* char_emb = (const float*)d_in[2];
    const float* word_emb = (const float*)d_in[3];
    const float* conv_w   = (const float*)d_in[4];
    const float* conv_b   = (const float*)d_in[5];
    const float* wih0     = (const float*)d_in[6];
    const float* whh0     = (const float*)d_in[7];
    const float* bih0     = (const float*)d_in[8];
    const float* bhh0     = (const float*)d_in[9];
    const float* wih1     = (const float*)d_in[10];
    const float* whh1     = (const float*)d_in[11];
    const float* bih1     = (const float*)d_in[12];
    const float* bhh1     = (const float*)d_in[13];
    const float* dense_w  = (const float*)d_in[14];
    const float* dense_b  = (const float*)d_in[15];
    float* out = (float*)d_out;

    // workspace carve-up
    char* p = (char*)d_ws;
    size_t off = 0;
    __bf16* emb   = (__bf16*)(p + off); off = alignup(off + (size_t)ROWS * INP0 * 2);
    __bf16* xg0   = (__bf16*)(p + off); off = alignup(off + (size_t)ROWS * NG * 2);
    __bf16* W0x   = (__bf16*)(p + off); off = alignup(off + (size_t)NG * INP0 * 2);
    __bf16* W0h   = (__bf16*)(p + off); off = alignup(off + (size_t)NG * HP * 2);
    __bf16* W1    = (__bf16*)(p + off); off = alignup(off + (size_t)NG * K1 * 2);
    __bf16* DW    = (__bf16*)(p + off); off = alignup(off + (size_t)64 * HP * 2);
    float*  bias0 = (float*) (p + off); off = alignup(off + NG * 4);
    float*  bias1 = (float*) (p + off); off = alignup(off + NG * 4);
    float*  DB    = (float*) (p + off); off = alignup(off + 64 * 4);
    __bf16* h1seq = (__bf16*)(p + off); off = alignup(off + (size_t)(SS + 1) * BB * HP * 2);
    __bf16* h2seq = (__bf16*)(p + off); off = alignup(off + (size_t)(SS + 1) * BB * HP * 2);
    float*  c0    = (float*) (p + off); off = alignup(off + (size_t)BB * HP * 4);
    float*  c1    = (float*) (p + off); off = alignup(off + (size_t)BB * HP * 4);

    prep_kernel<<<256, 256, 0, stream>>>(wih0, whh0, bih0, bhh0,
                                         wih1, whh1, bih1, bhh1,
                                         dense_w, dense_b,
                                         W0x, W0h, W1, DW, bias0, bias1, DB);
    zero_state<<<256, 256, 0, stream>>>(c0, c1, h1seq, h2seq);
    embed_kernel<<<ROWS, 128, 0, stream>>>(char_idx, word_idx, char_emb, word_emb,
                                           conv_w, conv_b, emb);
    input_proj<<<8192, 128, 0, stream>>>(emb, W0x, bias0, xg0);

    // sequential recurrence: 65 fused launches (layer0 step s || layer1 step s-1)
    for (int s = 0; s <= SS; ++s) {
        lstm_fused<<<256, 128, 0, stream>>>(s, xg0, h1seq, h2seq,
                                            W0h, W1, bias1, c0, c1);
    }
    dense_kernel<<<512, 128, 0, stream>>>(h2seq, DW, DB, out);
}
